// Feq_Cross_Correlation_11416023072815
// MI455X (gfx1250) — matrix operand order repaired and checked by reading, not compiled, not run on hardware
//
#include <hip/hip_runtime.h>
#include <hip/hip_bf16.h>
#include <math.h>

// ---------------- problem constants (from reference) ----------------
constexpr int B = 32, L = 2048, E = 127, NG = 8, I = 135;
constexpr int F = L / 2 + 1;        // 1025 rfft bins
constexpr int TOPK = 7;             // K-1 with K=8
constexpr int M = B * L;            // 65536 rows
constexpr float SCALE8 = 0.35355339059327373f;   // 1/sqrt(8)
constexpr float SCALE  = 0.08873565094161139f;   // 1/sqrt(127)

// ---------------- workspace layout (float offsets) ----------------
constexpr size_t O_VGP  = 0;                               // (M,8) values@gv_w+gv_b
constexpr size_t O_Q    = O_VGP + (size_t)M * NG;          // (M,) q projection
constexpr size_t O_SQ   = O_Q + M;                         // (B,)  sum_l q
constexpr size_t O_SKG  = O_SQ + B;                        // (B,8) sum_l keys_group
constexpr size_t O_WG   = O_SKG + B * NG;                  // (B,8) group softmax
constexpr size_t O_SK2  = O_WG + B * NG;                   // (B,I) sum_l k2
constexpr size_t O_KSUM = O_SK2 + B * I;                   // (M,)  sum_e k2
constexpr size_t O_V2   = O_KSUM + M;                      // (M,I) v2
constexpr size_t O_A    = O_V2 + (size_t)M * I;            // (B,F) |q_fft|
constexpr size_t O_AMAX = O_A + (size_t)B * F;             // (B,)
constexpr size_t O_C    = O_AMAX + B;                      // (B,F) |sum_e k_fft|
constexpr size_t O_VFS  = O_C + (size_t)B * F;             // (B,7,F,2) v_fft selected
constexpr size_t O_ZS   = O_VFS + (size_t)B * TOPK * F * 2;// (B,F) 1/(Z*I^2)
constexpr size_t O_VFT  = O_ZS + (size_t)B * F;            // (B,14,F) Vftt re/im rows
constexpr size_t O_VT   = O_VFT + (size_t)B * 2 * TOPK * F;// (M,7) irfft result
constexpr size_t O_W    = O_VT + (size_t)M * TOPK;         // (B,I) channel softmax
constexpr size_t O_WK   = O_W + (size_t)B * I;             // (7,)
constexpr size_t O_IDX  = O_WK + TOPK;                     // (7,) ints

constexpr size_t OUT1 = (size_t)M * (1 + I);          // 8,912,896
constexpr size_t OUT2 = (size_t)M * (1 + 2 * TOPK);   // 983,040

// ---------------- WMMA types ----------------
typedef __attribute__((ext_vector_type(16))) _Float16 v16h;
typedef __attribute__((ext_vector_type(8)))  float    v8f;

// ---------------- Tensor Data Mover (gfx1250) ----------------
#if defined(__HIP_DEVICE_COMPILE__) && defined(__gfx1250__) && \
    __has_builtin(__builtin_amdgcn_tensor_load_to_lds)
#define USE_TDM 1
#else
#define USE_TDM 0
#endif

#if USE_TDM
typedef __attribute__((ext_vector_type(4))) unsigned int u32x4;
typedef __attribute__((ext_vector_type(8))) int          i32x8;
typedef __attribute__((ext_vector_type(4))) int          i32x4;

// Issue one TDM descriptor: copy a (w x rows) fp32 tile (row stride
// `stride` elements) from global memory to contiguous LDS at lds_dst.
// D# per CDNA5 ISA §8.3/8.4: group0 {count=1, lds_addr, global_addr,
// type=2}; group1 {data_size=4B, tensor_dim0/1, tile_dim0/1, dim0 stride}.
// Toolchain uses the 6-arg builtin form (g0,g1,g2,g3,g4,cpol).
__device__ inline void tdm_load_f32(const float* gsrc, float* lds_dst,
                                    unsigned w, unsigned rows, unsigned stride) {
  const unsigned long long ga = (unsigned long long)(size_t)gsrc;
  const unsigned int la = (unsigned int)(size_t)lds_dst;  // addr[31:0] = LDS offset
  u32x4 g0 = { 1u,                                              // count=1
               la,                                              // lds_addr
               (unsigned int)(ga & 0xFFFFFFFFu),                // global_addr[31:0]
               (unsigned int)((ga >> 32) & 0x01FFFFFFu) | (2u << 30) }; // [56:32]|type=2
  i32x8 g1 = { (int)(2u << 16),                                 // data_size = 4 bytes
               (int)((w & 0xFFFFu) << 16),                      // tensor_dim0 lo16
               (int)(((w >> 16) & 0xFFFFu) | ((rows & 0xFFFFu) << 16)), // dim0 hi|dim1 lo
               (int)(((rows >> 16) & 0xFFFFu) | ((w & 0xFFFFu) << 16)), // dim1 hi|tile_dim0
               (int)(rows & 0xFFFFu),                           // tile_dim1 (tile_dim2=0)
               (int)stride,                                     // tensor_dim0_stride lo32
               0, 0 };
  i32x4 g2 = { 0, 0, 0, 0 };
  i32x4 g3 = { 0, 0, 0, 0 };
  i32x8 g4 = { 0, 0, 0, 0, 0, 0, 0, 0 };
  __builtin_amdgcn_tensor_load_to_lds(g0, g1, g2, g3, g4, 0);
}
#endif

// =====================================================================
// K1: q projection, group projections, and the only pieces of keys_group
//     we actually need (its per-batch sum; full values_groups kept).
// =====================================================================
__global__ void proj_group_kernel(const float* __restrict__ queries,
                                  const float* __restrict__ keys,
                                  const float* __restrict__ values,
                                  const float* __restrict__ gkw, const float* __restrict__ gkb,
                                  const float* __restrict__ gvw, const float* __restrict__ gvb,
                                  const float* __restrict__ qw,  const float* __restrict__ qb,
                                  float* __restrict__ qbuf, float* __restrict__ vgp,
                                  float* __restrict__ sq,   float* __restrict__ skg) {
  __shared__ float sgk[E * NG], sgv[E * NG];
  __shared__ float sacc[NG + 1];
  const int tid = threadIdx.x;
  for (int i = tid; i < E * NG; i += 256) { sgk[i] = gkw[i]; sgv[i] = gvw[i]; }
  if (tid < NG + 1) sacc[tid] = 0.f;
  __syncthreads();

  const int gid = blockIdx.x * 256 + tid;   // flattened (b,l); 256 | L
  const int b = gid >> 11;
  const float qv = queries[gid] * qw[0] + qb[0];
  qbuf[gid] = qv;

  float kg[NG] = {0.f}, vg[NG] = {0.f};
  const float* krow = keys + (size_t)gid * E;
  const float* vrow = values + (size_t)gid * E;
  __builtin_prefetch(krow, 0, 0);
  __builtin_prefetch(vrow, 0, 0);
  for (int k = 0; k < E; ++k) {
    const float kv = krow[k], vv = vrow[k];
#pragma unroll
    for (int g = 0; g < NG; ++g) {
      kg[g] += kv * sgk[k * NG + g];
      vg[g] += vv * sgv[k * NG + g];
    }
  }
#pragma unroll
  for (int g = 0; g < NG; ++g) {
    kg[g] += gkb[g];
    vgp[(size_t)gid * NG + g] = vg[g] + gvb[g];
    atomicAdd(&sacc[g], kg[g]);
  }
  atomicAdd(&sacc[NG], qv);
  __syncthreads();
  if (tid < NG)       atomicAdd(&skg[b * NG + tid], sacc[tid]);
  else if (tid == NG) atomicAdd(&sq[b], sacc[NG]);
}

// =====================================================================
// K2: group attention weights  w_g = softmax_g(Sq*Skg/L * scale8)
// =====================================================================
__global__ void group_softmax_kernel(const float* __restrict__ sq,
                                     const float* __restrict__ skg,
                                     float* __restrict__ wg) {
  const int b = threadIdx.x;
  if (b >= B) return;
  const float s = sq[b] * (1.0f / (float)L) * SCALE8;
  float v[NG], mx = -1e30f;
#pragma unroll
  for (int g = 0; g < NG; ++g) { v[g] = s * skg[b * NG + g]; mx = fmaxf(mx, v[g]); }
  float sum = 0.f;
#pragma unroll
  for (int g = 0; g < NG; ++g) sum += __expf(v[g] - mx);
#pragma unroll
  for (int g = 0; g < NG; ++g) wg[b * NG + g] = __expf(v[g] - mx) / sum;
}

// =====================================================================
// K3: WMMA projection GEMM.  Out[m,n] = sum_k X[m,k] * W[n,k] + bias[n]
//     X = [keys|values_g] (KMODE) or [values|values_g].
//     KMODE: k2 is never stored — only its row sums (ksum, FFT input)
//     and column sums (Sk2, for channel softmax / top-k).
//     Tile: 64(M)x16(N) per block, 4 waves, K padded 135->160 (5x32).
// =====================================================================
template <bool KMODE>
__global__ void proj_gemm(const float* __restrict__ X127,
                          const float* __restrict__ Wmat,
                          const float* __restrict__ bias,
                          const float* __restrict__ vgp,
                          const float* __restrict__ wg,
                          float* __restrict__ v2,
                          float* __restrict__ ksum,
                          float* __restrict__ sk2) {
  __shared__ _Float16 ldsA[64 * 160];   // 20 KB
  __shared__ _Float16 ldsB[160 * 16];   //  5 KB
  const int tid = threadIdx.x;          // 128 threads = 4 waves
  const int wave = tid >> 5, lane = tid & 31;
  const int m0 = blockIdx.x * 64;       // rows never cross a batch (64 | L)
  const int n0 = blockIdx.y * 16;
  const int bb = m0 >> 11;

  // stage B tile: W[n][k] -> ldsB[k][n]  (zero-padded)
  for (int i = tid; i < 160 * 16; i += 128) {
    const int k = i >> 4, n = i & 15, nG = n0 + n;
    float v = 0.f;
    if (nG < I && k < I) v = Wmat[nG * I + k];
    ldsB[k * 16 + n] = (_Float16)v;
  }
  // stage A tile: X built on the fly (keys/values | w_g * values_groups)
  for (int i = tid; i < 64 * 160; i += 128) {
    const int r = i / 160, k = i - r * 160;
    const int gm = m0 + r;
    float v = 0.f;
    if (k < E)      v = X127[(size_t)gm * E + k];
    else if (k < I) v = wg[bb * NG + (k - E)] * vgp[(size_t)gm * NG + (k - E)];
    ldsA[r * 160 + k] = (_Float16)v;
  }
  __syncthreads();

  const int mw = wave * 16;
  v8f acc = {};
  for (int kk = 0; kk < 5; ++kk) {
    const int k0 = kk * 32;
    v16h af, bf;
    const int mrow = mw + (lane & 15);
#pragma unroll
    for (int h = 0; h < 16; ++h) {        // A: 16-bit 16x32 layout (ISA 7.12.2)
      const int r = h >> 1, p = h & 1;
      const int g = ((r >= 4) ? 2 : 0) + ((lane >= 16) ? 1 : 0);
      af[h] = ldsA[mrow * 160 + k0 + 8 * g + 2 * (r & 3) + p];
    }
#pragma unroll
    for (int h = 0; h < 16; ++h)          // B: N=lane&15, K in same order as A
      bf[h] = ldsB[(k0 + ((h >> 3) << 4) + 8 * (lane >> 4) + (h & 7)) * 16 + (lane & 15)];
    acc = __builtin_amdgcn_wmma_f32_16x16x32_f16(false, af, false, bf,
                                                 (short)0, acc, false, false);
  }

  // epilogue: C/D layout lanes 0-15 -> M=r, lanes 16-31 -> M=r+8, N=lane&15
  const int n = lane & 15, nG = n0 + n;
  const int mBase = (lane >= 16) ? 8 : 0;
  const float bv = (nG < I) ? bias[nG] : 0.f;
  float cb[8];
#pragma unroll
  for (int r = 0; r < 8; ++r) cb[r] = (nG < I) ? (acc[r] + bv) : 0.f;

  if (!KMODE) {
    if (nG < I) {
#pragma unroll
      for (int r = 0; r < 8; ++r)
        v2[(size_t)(m0 + mw + mBase + r) * I + nG] = cb[r];
    }
  } else {
    // column sums over the 16 M rows -> Sk2[b, n]
    float cs = 0.f;
#pragma unroll
    for (int r = 0; r < 8; ++r) cs += cb[r];
    cs += __shfl_xor(cs, 16);
    if (lane < 16 && nG < I) atomicAdd(&sk2[bb * I + nG], cs);
    // row sums over the 16 N cols -> ksum[b*L + l]
#pragma unroll
    for (int r = 0; r < 8; ++r) {
      float rs = cb[r];
      rs += __shfl_xor(rs, 1);
      rs += __shfl_xor(rs, 2);
      rs += __shfl_xor(rs, 4);
      rs += __shfl_xor(rs, 8);
      if ((lane & 15) == 0) atomicAdd(&ksum[m0 + mw + mBase + r], rs);
    }
  }
}

// =====================================================================
// K4: channel stats.  corr_mean[b,e] = Sq[b]*Sk2[b,e]/L  (DC-bin identity)
//     -> weight softmax, batch-mean top-k channels, weight_k softmax.
// =====================================================================
__global__ void stats_kernel(const float* __restrict__ sq,
                             const float* __restrict__ sk2,
                             float* __restrict__ weight,
                             float* __restrict__ wk,
                             int* __restrict__ idxArr,
                             float* __restrict__ out_tail) {
  __shared__ float cm[B * I];
  __shared__ float cmm[I];
  const int tid = threadIdx.x;
  for (int i = tid; i < B * I; i += 256)
    cm[i] = sq[i / I] * sk2[i] * (1.0f / (float)L);
  __syncthreads();
  if (tid < B) {  // per-batch softmax over channels
    const int b = tid;
    float mx = -1e30f;
    for (int e = 0; e < I; ++e) mx = fmaxf(mx, cm[b * I + e] * SCALE);
    float s = 0.f;
    for (int e = 0; e < I; ++e) s += __expf(cm[b * I + e] * SCALE - mx);
    const float inv = 1.0f / s;
    for (int e = 0; e < I; ++e)
      weight[b * I + e] = __expf(cm[b * I + e] * SCALE - mx) * inv;
  }
  if (tid < I) {  // batch means
    float s = 0.f;
    for (int b = 0; b < B; ++b) s += cm[b * I + tid];
    cmm[tid] = s * (1.0f / (float)B);
  }
  __syncthreads();
  if (tid == 0) {  // top-7 selection + weight_k softmax
    float tv[TOPK]; int ti[TOPK];
    bool used[I];
    for (int e = 0; e < I; ++e) used[e] = false;
    for (int k = 0; k < TOPK; ++k) {
      float best = -1e30f; int bi = 0;
      for (int e = 0; e < I; ++e)
        if (!used[e] && cmm[e] > best) { best = cmm[e]; bi = e; }
      used[bi] = true; tv[k] = best; ti[k] = bi;
    }
    float mx = -1e30f;
    for (int k = 0; k < TOPK; ++k) mx = fmaxf(mx, tv[k] * SCALE);
    float s = 0.f;
    for (int k = 0; k < TOPK; ++k) s += __expf(tv[k] * SCALE - mx);
    for (int k = 0; k < TOPK; ++k) {
      const float w = __expf(tv[k] * SCALE - mx) / s;
      wk[k] = w; idxArr[k] = ti[k]; out_tail[k] = w;
    }
  }
}

// =====================================================================
// LDS radix-2 FFT, length 2048, 256 threads per row.
// =====================================================================
__device__ void block_fft(float* re, float* im, int inverse) {
  const int tid = threadIdx.x;
  for (int i = tid; i < 2048; i += 256) {
    const int j = (int)(__brev((unsigned)i) >> (32 - 11));
    if (j > i) {
      float tr = re[i]; re[i] = re[j]; re[j] = tr;
      float ti = im[i]; im[i] = im[j]; im[j] = ti;
    }
  }
  __syncthreads();
  for (int len = 2; len <= 2048; len <<= 1) {
    const int half = len >> 1;
    const float ang0 = (inverse ? 2.0f : -2.0f) * 3.14159265358979323846f / (float)len;
    for (int idx = tid; idx < 1024; idx += 256) {
      const int blk = idx / half, pos = idx - blk * half;
      const int i = blk * len + pos, j = i + half;
      float s, c;
      __sincosf(ang0 * (float)pos, &s, &c);
      const float xr = re[j] * c - im[j] * s;
      const float xi = re[j] * s + im[j] * c;
      const float ur = re[i], ui = im[i];
      re[i] = ur + xr; im[i] = ui + xi;
      re[j] = ur - xr; im[j] = ui - xi;
    }
    __syncthreads();
  }
}

// =====================================================================
// K5: forward FFTs. rows 0..31: q -> |q_fft| (+max); 32..63: sum_e k2
//     -> |Ks|; 64..287: the 7 selected v2 channels -> complex v_fft.
//     Row data is DMA'd global->LDS by the Tensor Data Mover (wave 0
//     issues the descriptor, waits on TENSORcnt, workgroup barrier
//     publishes the LDS tile to the other waves).
// =====================================================================
__global__ void fft_forward(const float* __restrict__ qbuf,
                            const float* __restrict__ ksum,
                            const float* __restrict__ v2,
                            const int* __restrict__ idxArr,
                            float* __restrict__ amag, float* __restrict__ amax,
                            float* __restrict__ cmag, float* __restrict__ vfsel) {
  __shared__ float sre[2048], sim[2048], red[256];
  const int tid = threadIdx.x;
  const int row = blockIdx.x;
#if USE_TDM
  if (tid < 32) {  // wave 0 issues the tensor DMA (EXEC-independent)
    if (row < B) {
      tdm_load_f32(qbuf + (size_t)row * L, sre, L, 1, L);
    } else if (row < 2 * B) {
      tdm_load_f32(ksum + (size_t)(row - B) * L, sre, L, 1, L);
    } else {
      const int t = row - 2 * B, b = t / TOPK, j = t - b * TOPK;
      // 1-wide x 2048-row tile, row stride I: gathers one v2 channel
      tdm_load_f32(v2 + (size_t)b * L * I + idxArr[j], sre, 1, L, I);
    }
    __builtin_amdgcn_s_wait_tensorcnt(0);
  }
  for (int i = tid; i < 2048; i += 256) sim[i] = 0.f;
#else
  if (row < B) {
    for (int i = tid; i < 2048; i += 256) { sre[i] = qbuf[row * L + i]; sim[i] = 0.f; }
  } else if (row < 2 * B) {
    const int b = row - B;
    for (int i = tid; i < 2048; i += 256) { sre[i] = ksum[b * L + i]; sim[i] = 0.f; }
  } else {
    const int t = row - 2 * B, b = t / TOPK, j = t - b * TOPK;
    const int ch = idxArr[j];
    for (int i = tid; i < 2048; i += 256) {
      sre[i] = v2[((size_t)b * L + i) * I + ch];
      sim[i] = 0.f;
    }
  }
#endif
  __syncthreads();
  block_fft(sre, sim, 0);
  if (row < B) {
    const int b = row;
    float mx = 0.f;
    for (int l = tid; l < F; l += 256) {
      const float m = sqrtf(sre[l] * sre[l] + sim[l] * sim[l]);
      amag[(size_t)b * F + l] = m;
      mx = fmaxf(mx, m);
    }
    red[tid] = mx;
    __syncthreads();
    for (int s = 128; s > 0; s >>= 1) {
      if (tid < s) red[tid] = fmaxf(red[tid], red[tid + s]);
      __syncthreads();
    }
    if (tid == 0) amax[b] = red[0];
  } else if (row < 2 * B) {
    const int b = row - B;
    for (int s = tid; s < F; s += 256)
      cmag[(size_t)b * F + s] = sqrtf(sre[s] * sre[s] + sim[s] * sim[s]);
  } else {
    const int t = row - 2 * B, b = t / TOPK, j = t - b * TOPK;
    float* dst = vfsel + (((size_t)b * TOPK + j) * F) * 2;
    for (int s = tid; s < F; s += 256) { dst[2 * s] = sre[s]; dst[2 * s + 1] = sim[s]; }
  }
}

// =====================================================================
// K6: stores zinv_s = 1/(I^2 * Z_s), Z_s = sum_l exp(c_s*(a_l - a_max)).
//     (Z >= 1, so the reciprocal is safe; folding 1/I^2 here removes
//     the division from vftt_gemm's inner staging loop.)
// =====================================================================
__global__ void z_kernel(const float* __restrict__ amag,
                         const float* __restrict__ amax,
                         const float* __restrict__ cmag,
                         float* __restrict__ zinv) {
  __shared__ float sa[F];
  const int tid = threadIdx.x;
  const int b = blockIdx.y;
  for (int l = tid; l < F; l += 256) sa[l] = amag[(size_t)b * F + l];
  __syncthreads();
  const int s = blockIdx.x * 256 + tid;
  if (s < F) {
    const float cs = cmag[(size_t)b * F + s], amx = amax[b];
    float z = 0.f;
    for (int l = 0; l < F; ++l) z += __expf(cs * (sa[l] - amx));
    zinv[(size_t)b * F + s] = 1.0f / ((float)(I * I) * z);
  }
}

// =====================================================================
// K7: Vftt WMMA GEMM.  Vftt[b,m,l] = sum_s (vfsel*zinv)[b,m,s] *
//     exp(c_s*(a_l - a_max)).  Softmax-weight tile generated on the fly
//     in LDS — the (B,F,F) qk tensor is never materialized anywhere.
//     Per-tile invariants (a_l - a_max) hoisted; per-step c_s / zinv
//     staged once per chunk.  Rows m: 7 channels x {re,im} pad to 16.
// =====================================================================
__global__ void vftt_gemm(const float* __restrict__ vfsel,
                          const float* __restrict__ zinv,
                          const float* __restrict__ cmag,
                          const float* __restrict__ amag,
                          const float* __restrict__ amax,
                          float* __restrict__ vftt) {
  __shared__ _Float16 lA[16 * 32];
  __shared__ _Float16 lW[32 * 16];
  __shared__ float sAl[16], sCs[32], sZ[32];
  const int lane = threadIdx.x;           // 32 threads = 1 wave
  const int n0 = blockIdx.x * 16;
  const int b = blockIdx.y;
  const float amx = amax[b];

  if (lane < 16) {                        // loop-invariant (a_l - a_max)
    const int l = n0 + lane;
    sAl[lane] = (l < F) ? (amag[(size_t)b * F + l] - amx) : 0.f;
  }
  __syncthreads();

  v8f acc = {};
  for (int kk = 0; kk < 33; ++kk) {
    const int s0 = kk * 32;
    const int sl = s0 + lane;
    __syncthreads();
    sCs[lane] = (sl < F) ? cmag[(size_t)b * F + sl] : 0.f;
    sZ[lane]  = (sl < F) ? zinv[(size_t)b * F + sl] : 0.f;
    __syncthreads();
    for (int i = lane; i < 16 * 32; i += 32) {
      const int m = i >> 5, sk = i & 31, s = s0 + sk;
      float v = 0.f;
      if (m < 2 * TOPK && s < F)
        v = vfsel[(((size_t)b * TOPK + (m >> 1)) * F + s) * 2 + (m & 1)] * sZ[sk];
      lA[m * 32 + sk] = (_Float16)v;
    }
    for (int i = lane; i < 32 * 16; i += 32) {
      const int sk = i >> 4, n = i & 15;
      float v = 0.f;
      if (s0 + sk < F && n0 + n < F) v = __expf(sCs[sk] * sAl[n]);
      lW[sk * 16 + n] = (_Float16)v;
    }
    __syncthreads();
    v16h af, bf;
    const int mrow = lane & 15;
#pragma unroll
    for (int h = 0; h < 16; ++h) {
      const int r = h >> 1, p = h & 1;
      const int g = ((r >= 4) ? 2 : 0) + ((lane >= 16) ? 1 : 0);
      af[h] = lA[mrow * 32 + 8 * g + 2 * (r & 3) + p];
    }
#pragma unroll
    for (int h = 0; h < 16; ++h) bf[h] = lW[(((h >> 3) << 4) + 8 * (lane >> 4) + (h & 7)) * 16 + (lane & 15)];
    acc = __builtin_amdgcn_wmma_f32_16x16x32_f16(false, af, false, bf,
                                                 (short)0, acc, false, false);
  }
  const int n = lane & 15, l = n0 + n;
  const int mBase = (lane >= 16) ? 8 : 0;
  if (l < F) {
#pragma unroll
    for (int r = 0; r < 8; ++r) {
      const int m = mBase + r;
      if (m < 2 * TOPK) vftt[((size_t)b * (2 * TOPK) + m) * F + l] = acc[r];
    }
  }
}

// =====================================================================
// K8: irfft of the 7 selected Vftt channels (Hermitian-extended).
// =====================================================================
__global__ void fft_inverse(const float* __restrict__ vftt,
                            float* __restrict__ vtime) {
  __shared__ float sre[2048], sim[2048];
  const int tid = threadIdx.x, t = blockIdx.x;
  const int b = t / TOPK, j = t - b * TOPK;
  const float* hre = vftt + ((size_t)b * (2 * TOPK) + 2 * j) * F;
  const float* him = vftt + ((size_t)b * (2 * TOPK) + 2 * j + 1) * F;
  for (int i = tid; i < 2048; i += 256) {
    if (i <= 1024) { sre[i] = hre[i];        sim[i] = him[i]; }
    else           { sre[i] = hre[2048 - i]; sim[i] = -him[2048 - i]; }
  }
  __syncthreads();
  block_fft(sre, sim, 1);
  const float inv = 1.0f / 2048.0f;
  for (int n = tid; n < 2048; n += 256)
    vtime[((size_t)b * L + n) * TOPK + j] = sre[n] * inv;
}

// =====================================================================
// K9: output assembly.
// =====================================================================
__global__ void assemble_main(const float* __restrict__ queries,
                              const float* __restrict__ weight,
                              const float* __restrict__ v2,
                              float* __restrict__ out) {
  const size_t t = (size_t)blockIdx.x * 256 + threadIdx.x;
  if (t >= OUT1) return;
  const size_t row = t / (1 + I);
  const int c = (int)(t - row * (1 + I));
  const int b = (int)(row >> 11);
  float v;
  if (c == 0) v = queries[row];
  else        v = weight[b * I + (c - 1)] * v2[row * I + (c - 1)];
  out[t] = v;
}

__global__ void assemble_topk(const float* __restrict__ queries,
                              const float* __restrict__ wk,
                              const int* __restrict__ idxArr,
                              const float* __restrict__ v2,
                              const float* __restrict__ vtime,
                              float* __restrict__ out2) {
  constexpr int C = 1 + 2 * TOPK;
  const size_t t = (size_t)blockIdx.x * 256 + threadIdx.x;
  if (t >= OUT2) return;
  const size_t row = t / C;
  const int c = (int)(t - row * C);
  float v;
  if (c == 0)          v = queries[row];
  else if (c <= TOPK) { const int j = c - 1;        v = wk[j] * v2[row * I + idxArr[j]]; }
  else               { const int j = c - 1 - TOPK;  v = wk[j] * vtime[row * TOPK + j]; }
  out2[t] = v;
}

// =====================================================================
// launcher
// =====================================================================
extern "C" void kernel_launch(void* const* d_in, const int* in_sizes, int n_in,
                              void* d_out, int out_size, void* d_ws, size_t ws_size,
                              hipStream_t stream) {
  (void)in_sizes; (void)n_in; (void)out_size; (void)ws_size;
  const float* queries = (const float*)d_in[0];
  const float* keys    = (const float*)d_in[1];
  const float* values  = (const float*)d_in[2];
  const float* gkw     = (const float*)d_in[3];
  const float* gkb     = (const float*)d_in[4];
  const float* gvw     = (const float*)d_in[5];
  const float* gvb     = (const float*)d_in[6];
  const float* qw      = (const float*)d_in[7];
  const float* qb      = (const float*)d_in[8];
  const float* kw      = (const float*)d_in[9];
  const float* kb      = (const float*)d_in[10];
  const float* vw      = (const float*)d_in[11];
  const float* vb      = (const float*)d_in[12];
  float* ws  = (float*)d_ws;
  float* out = (float*)d_out;

  float* vgp  = ws + O_VGP;  float* qbuf = ws + O_Q;
  float* sq   = ws + O_SQ;   float* skg  = ws + O_SKG;
  float* wg   = ws + O_WG;   float* sk2  = ws + O_SK2;
  float* ksum = ws + O_KSUM; float* v2   = ws + O_V2;
  float* amag = ws + O_A;    float* amax = ws + O_AMAX;
  float* cmag = ws + O_C;    float* vfsel = ws + O_VFS;
  float* zinv = ws + O_ZS;   float* vftt = ws + O_VFT;
  float* vtime = ws + O_VT;  float* weight = ws + O_W;
  float* wkv  = ws + O_WK;
  int*   idxArr = (int*)(ws + O_IDX);

  // zero all atomically-accumulated scratch every call (deterministic work)
  (void)hipMemsetAsync(ws + O_SQ, 0, (O_V2 - O_SQ) * sizeof(float), stream);

  proj_group_kernel<<<M / 256, 256, 0, stream>>>(queries, keys, values, gkw, gkb,
                                                 gvw, gvb, qw, qb, qbuf, vgp, sq, skg);
  group_softmax_kernel<<<1, 32, 0, stream>>>(sq, skg, wg);
  proj_gemm<true ><<<dim3(M / 64, 9), 128, 0, stream>>>(keys,   kw, kb, vgp, wg, v2, ksum, sk2);
  proj_gemm<false><<<dim3(M / 64, 9), 128, 0, stream>>>(values, vw, vb, vgp, wg, v2, ksum, sk2);
  stats_kernel<<<1, 256, 0, stream>>>(sq, sk2, weight, wkv, idxArr, out + OUT1 + OUT2);
  fft_forward<<<2 * B + B * TOPK, 256, 0, stream>>>(qbuf, ksum, v2, idxArr,
                                                    amag, amax, cmag, vfsel);
  z_kernel<<<dim3((F + 255) / 256, B), 256, 0, stream>>>(amag, amax, cmag, zinv);
  vftt_gemm<<<dim3((F + 15) / 16, B), 32, 0, stream>>>(vfsel, zinv, cmag, amag, amax, vftt);
  fft_inverse<<<B * TOPK, 256, 0, stream>>>(vftt, vtime);
  assemble_main<<<(unsigned)((OUT1 + 255) / 256), 256, 0, stream>>>(queries, weight, v2, out);
  assemble_topk<<<(unsigned)((OUT2 + 255) / 256), 256, 0, stream>>>(queries, wkv, idxArr,
                                                                    v2, vtime, out + OUT1);
}
